// STU_14104672600702
// MI455X (gfx1250) — matrix operand order repaired and checked by reading, not compiled, not run on hardware
//
#include <hip/hip_runtime.h>

// ---------------- CDNA5 WMMA types ----------------
typedef __attribute__((ext_vector_type(16))) __bf16 v16bf;
typedef __attribute__((ext_vector_type(8)))  float  v8f;

#define B_BATCH 2
#define L_SEQ   4096
#define D_DIM   128
#define K_FILT  16
#define E_DIM   128
#define LT      16        // output rows per workgroup (WMMA M)
#define TC      32        // t-chunk size (WMMA K for bf16)
#define NWAVES  8
#define FREV    4160      // reversed-filter row stride (zero-padded past 4096)

union FragU {
    v16bf v;
    unsigned short h[16];
    unsigned u[8];
    uint4 q[2];
};

// A-matrix fragment (16x32 bf16, M x K). lane%16 = M row;
// half-wave 0: K 0..15; half-wave 1: K 16..31 (matches B fragment).
// p points at this lane's M-row: p[kk] = A(m, kk). Unaligned-safe; the
// backend merges these into (unaligned) global_load_b128 pairs.
__device__ __forceinline__ v16bf load_fragA(const __bf16* p, int lane) {
    const unsigned short* ps = (const unsigned short*)p;
    const int kb = (lane >> 4) << 4;
    FragU r;
#pragma unroll
    for (int j = 0; j < 4; ++j) {
        r.h[2 * j]     = ps[kb + 2 * j];
        r.h[2 * j + 1] = ps[kb + 2 * j + 1];
    }
#pragma unroll
    for (int j = 0; j < 4; ++j) {
        r.h[8 + 2 * j]     = ps[kb + 8 + 2 * j];
        r.h[8 + 2 * j + 1] = ps[kb + 8 + 2 * j + 1];
    }
    return r.v;
}

// B-matrix fragment (32x16 bf16, K x N), contiguous k order used by this code (matches load_fragA) for
// K=32: lanes 0-15 hold K 0..15, lanes 16-31 hold K 16..31; N = lane%16.
// p points at this lane's N-column (p[kk] = B(kk,n)) and p is 32B-aligned,
// so the 16 contiguous halfs are two global_load_b128.
__device__ __forceinline__ v16bf load_fragB_aligned(const __bf16* p, int lane) {
    const uint4* q = (const uint4*)(p + ((lane >> 4) << 4));
    FragU r;
    r.q[0] = q[0];
    r.q[1] = q[1];
    return r.v;
}

__device__ __forceinline__ v8f wmma_bf16(v16bf a, v16bf b, v8f c) {
    return __builtin_amdgcn_wmma_f32_16x16x32_bf16(
        false, a, false, b, (short)0, c, false, false);
}

// ---------------- prep kernels (one-time conversions) ----------------
// Mt[sign][k][e][d] = M_sign[k][d][e] as bf16 (B-fragment friendly: d contiguous)
__global__ void stu_prep_Mt(const float* __restrict__ Mp,
                            const float* __restrict__ Mm,
                            __bf16* __restrict__ Mt) {
    int i = blockIdx.x * blockDim.x + threadIdx.x;  // 2*16*128*128
    if (i >= 2 * K_FILT * E_DIM * D_DIM) return;
    int d = i & 127, e = (i >> 7) & 127, k = (i >> 14) & 15, s = (i >> 18) & 1;
    const float* M = s ? Mm : Mp;
    Mt[i] = (__bf16)M[(k * D_DIM + d) * E_DIM + e];
}

// xTg[b][d][t] = x[b][t][d] as bf16 (t contiguous -> aligned B fragments)
__global__ void stu_prep_xT(const float* __restrict__ x,
                            __bf16* __restrict__ xTg) {
    int i = blockIdx.x * blockDim.x + threadIdx.x;  // 2*128*4096
    if (i >= B_BATCH * D_DIM * L_SEQ) return;
    int t = i & 4095, d = (i >> 12) & 127, b = i >> 19;
    xTg[i] = (__bf16)x[((size_t)b * L_SEQ + t) * D_DIM + d];
}

// filtRev[0][k][j] = phi_k[4095-j], filtRev[1][k][j] = (-1)^(4095-j) phi_k[4095-j]
// (psi = sign-modulated filter of the "negative featurization" branch),
// rows zero-padded to FREV so Toeplitz fragments never go out of bounds.
__global__ void stu_prep_filtRev(const float* __restrict__ filt,
                                 __bf16* __restrict__ filtRev) {
    int i = blockIdx.x * blockDim.x + threadIdx.x;  // 2*16*FREV
    if (i >= 2 * K_FILT * FREV) return;
    int j  = i % FREV;
    int k  = (i / FREV) % K_FILT;
    int sg = i / (K_FILT * FREV);
    float v = 0.0f;
    if (j < L_SEQ) {
        int s = L_SEQ - 1 - j;
        v = filt[s * K_FILT + k];
        if (sg && (s & 1)) v = -v;
    }
    filtRev[i] = (__bf16)v;
}

// ---------------- main fused STU kernel ----------------
// Grid: 512 blocks (256 l-tiles x 2 batches), 256 threads (8 wave32s).
// Each wave owns 2 k-filters x both signs.
__global__ __launch_bounds__(256, 1) void stu_main(
    const __bf16* __restrict__ xTg,      // (B, D, L) bf16
    const __bf16* __restrict__ filtRev,  // (2, K, FREV) bf16 reversed phi/psi
    const __bf16* __restrict__ Mt,       // (2, K, E, D) bf16
    float* __restrict__ out)             // (B, L, E)
{
    constexpr int UPAD = 40;   // U-stage row stride (halfs)
    constexpr int EPAD = 132;  // out-reduce row stride (floats)

    __shared__ unsigned short uStage[NWAVES * 16 * UPAD];
    __shared__ float outRed[16 * EPAD];

    const int tid  = threadIdx.x;
    const int wave = tid >> 5;
    const int lane = tid & 31;
    const int n16  = lane & 15;

    const int b  = blockIdx.x >> 8;          // 256 l-tiles per batch
    const int l0 = (blockIdx.x & 255) << 4;

    for (int i = tid; i < 16 * EPAD; i += 256) outRed[i] = 0.0f;

    const v8f vzero = {};
    v8f oacc[8];                             // 16 x 128 output partials
#pragma unroll
    for (int eb = 0; eb < 8; ++eb) oacc[eb] = vzero;

    const int nChunks = (l0 + LT - 1) / TC + 1;   // t0 = 0 .. l0+15, step 32

    for (int dp = 0; dp < 4; ++dp) {              // 4 d-pairs of 32 columns
        const int d0 = dp * 32;

        // per-lane x rows for the two 16-wide d-blocks of this d-pair
        const __bf16* xrow0 = xTg + ((size_t)(b * D_DIM + d0 + n16) * L_SEQ);
        const __bf16* xrow1 = xrow0 + (size_t)16 * L_SEQ;

        v8f acc[4][2];   // p = 2*kk2 + sign; x 2 d-blocks
#pragma unroll
        for (int p = 0; p < 4; ++p) { acc[p][0] = vzero; acc[p][1] = vzero; }

        for (int ch = 0; ch < nChunks; ++ch) {
            const int t0 = ch * TC;
            const int c  = l0 - t0;           // A(m,kk) = phi_k[c + m - kk]

            v16bf bx0 = load_fragB_aligned(xrow0 + t0, lane);
            v16bf bx1 = load_fragB_aligned(xrow1 + t0, lane);
            if (t0 + TC < L_SEQ)
                __builtin_prefetch(xrow0 + t0 + TC, 0, 1);

            // A(m,kk) = filtRev[sg][k][(4095 - c - m) + kk]: contiguous in kk
            const int arow = (L_SEQ - 1) - c - n16;   // >= 0 always

#pragma unroll
            for (int kk2 = 0; kk2 < 2; ++kk2) {
                const int k = wave * 2 + kk2;
                const __bf16* pa = filtRev + (size_t)k * FREV + arow;
                v16bf aP = load_fragA(pa, lane);                            // phi
                v16bf aM = load_fragA(pa + (size_t)K_FILT * FREV, lane);    // psi
                acc[2 * kk2 + 0][0] = wmma_bf16(aP, bx0, acc[2 * kk2 + 0][0]);
                acc[2 * kk2 + 0][1] = wmma_bf16(aP, bx1, acc[2 * kk2 + 0][1]);
                acc[2 * kk2 + 1][0] = wmma_bf16(aM, bx0, acc[2 * kk2 + 1][0]);
                acc[2 * kk2 + 1][1] = wmma_bf16(aM, bx1, acc[2 * kk2 + 1][1]);
            }
        }

        // ---- stage 2: oacc[16 x E] += U(16 x 32) x Mt(32 x 16) per e-block ----
        unsigned short* myU = &uStage[wave * 16 * UPAD];
        const int mrow = ((lane >> 4) << 3);  // C-layout: M = r + 8*(lane/16)
#pragma unroll
        for (int p = 0; p < 4; ++p) {
            const int k  = wave * 2 + (p >> 1);
            const int sg = p & 1;
            __syncthreads();                  // uniform; orders reuse of myU
#pragma unroll
            for (int db = 0; db < 2; ++db)
#pragma unroll
                for (int r = 0; r < 8; ++r) {
                    union { __bf16 b; unsigned short s; } cv;
                    cv.b = (__bf16)acc[p][db][r];
                    myU[(mrow + r) * UPAD + db * 16 + n16] = cv.s;
                }
            __syncthreads();

            v16bf ua = load_fragA((const __bf16*)&myU[n16 * UPAD], lane);
#pragma unroll
            for (int eb = 0; eb < 8; ++eb) {
                const __bf16* pm =
                    Mt + (((size_t)sg * K_FILT + k) * E_DIM + (eb * 16 + n16)) * D_DIM + d0;
                v16bf mb = load_fragB_aligned(pm, lane);
                oacc[eb] = wmma_bf16(ua, mb, oacc[eb]);
            }
        }
    }

    // ---- cross-wave reduction (ds_add_f32) and store ----
    __syncthreads();
    {
        const int mrow = ((lane >> 4) << 3);
#pragma unroll
        for (int eb = 0; eb < 8; ++eb)
#pragma unroll
            for (int r = 0; r < 8; ++r)
                atomicAdd(&outRed[(mrow + r) * EPAD + eb * 16 + n16], oacc[eb][r]);
    }
    __syncthreads();

    for (int i = tid; i < 16 * E_DIM; i += 256) {
        int m = i >> 7, e = i & 127;
        out[((size_t)b * L_SEQ + (l0 + m)) * E_DIM + e] = outRed[m * EPAD + e];
    }
}

extern "C" void kernel_launch(void* const* d_in, const int* in_sizes, int n_in,
                              void* d_out, int out_size, void* d_ws, size_t ws_size,
                              hipStream_t stream) {
    (void)in_sizes; (void)n_in; (void)out_size; (void)ws_size;
    const float* x    = (const float*)d_in[0];   // (2, 4096, 128)
    const float* filt = (const float*)d_in[1];   // (4096, 16)
    const float* Mp   = (const float*)d_in[2];   // (16, 128, 128)
    const float* Mm   = (const float*)d_in[3];   // (16, 128, 128)
    float* out = (float*)d_out;                  // (2, 4096, 128)

    // workspace layout (bf16):
    //   Mt      : 2*16*128*128 = 1 MiB     @ 0
    //   xTg     : 2*128*4096   = 2 MiB     @ 1 MiB
    //   filtRev : 2*16*4160    = 260 KiB   @ 3 MiB
    char* ws = (char*)d_ws;
    __bf16* Mt      = (__bf16*)(ws);
    __bf16* xTg     = (__bf16*)(ws + (1u << 20));
    __bf16* filtRev = (__bf16*)(ws + (3u << 20));

    const int nM = 2 * K_FILT * E_DIM * D_DIM;
    stu_prep_Mt<<<(nM + 255) / 256, 256, 0, stream>>>(Mp, Mm, Mt);
    const int nX = B_BATCH * D_DIM * L_SEQ;
    stu_prep_xT<<<(nX + 255) / 256, 256, 0, stream>>>(x, xTg);
    const int nF = 2 * K_FILT * FREV;
    stu_prep_filtRev<<<(nF + 255) / 256, 256, 0, stream>>>(filt, filtRev);

    stu_main<<<B_BATCH * (L_SEQ / LT), 256, 0, stream>>>(xTg, filtRev, Mt, out);
}
